// SNNLayer_60894046323308
// MI455X (gfx1250) — matrix operand order repaired and checked by reading, not compiled, not run on hardware
//
#include <hip/hip_runtime.h>
#include <stdint.h>

typedef __bf16 v16bf __attribute__((ext_vector_type(16)));
typedef float  v8f   __attribute__((ext_vector_type(8)));
typedef float  f32x4 __attribute__((ext_vector_type(4)));
typedef int    i32x4 __attribute__((ext_vector_type(4)));
typedef int    i32x8 __attribute__((ext_vector_type(8)));
typedef unsigned int u32x4 __attribute__((ext_vector_type(4)));

#define T_DIM 256
#define I_DIM 1024
#define O_DIM 1024
#define B_DIM 128
#define OTILE 128
#define KC    32
#define NKC   (I_DIM / KC)   // 32 K-chunks
#define NTB   (T_DIM / 16)   // 16 t-blocks

// LDS layout (bytes), rows padded against bank conflicts
#define XROW_H   40                       // halfwords per xs row (32 data + 8 pad)
#define XS_BYTES (T_DIM * XROW_H * 2)     // 20480
#define WROW_F   36                       // floats per ws row (32 data + 4 TDM pad)
#define WS_OFF   XS_BYTES
#define WS_BYTES (OTILE * WROW_F * 4)     // 18432 (one buffer; x2 for double buffer)
#define Z_OFF    (WS_OFF + 2 * WS_BYTES)  // 57344
#define ZROW_F   260                      // 256 t + 4 pad floats (stride = 4 banks)
#define Z_BYTES  (OTILE * ZROW_F * 4)     // 133120
#define LDS_TOTAL (Z_OFF + Z_BYTES)       // 190464 < 320K

__device__ __forceinline__ unsigned pack_bf16x2(float lo, float hi) {
  union { __bf16 h[2]; unsigned u; } p;
  p.h[0] = (__bf16)lo;                    // native f32->bf16 (v_cvt_pk_bf16_f32)
  p.h[1] = (__bf16)hi;
  return p.u;
}

// TDM: W chunk [128 o][32 i] f32 -> LDS, pad 4 dwords every 32 (rows -> 36 floats)
__device__ __forceinline__ void tdm_load_w(const float* src, unsigned ldsAddr) {
  unsigned long long ga = (unsigned long long)(const void*)src;
  u32x4 g0;
  g0[0] = 1u;                                  // count=1, user descriptor
  g0[1] = ldsAddr;                             // lds_addr
  g0[2] = (unsigned int)(ga & 0xFFFFFFFFull);  // global_addr lo
  g0[3] = (unsigned int)((ga >> 32) & 0x01FFFFFFull) | (2u << 30); // type=2
  i32x8 g1;
  g1[0] = (2 << 16)      // data_size = 4B
        | (1 << 20)      // pad_enable
        | (4 << 22)      // pad_interval: every 32 dwords
        | (3 << 25);     // pad_amount: 4 dwords
  g1[1] = (int)((I_DIM & 0xFFFF) << 16);                              // tensor_dim0 lo
  g1[2] = (int)(((I_DIM >> 16) & 0xFFFF) | ((O_DIM & 0xFFFF) << 16)); // dim0 hi|dim1 lo
  g1[3] = (int)(((O_DIM >> 16) & 0xFFFF) | (KC << 16));               // dim1 hi|tile_dim0
  g1[4] = (int)(OTILE & 0xFFFF);                                      // tile_dim1
  g1[5] = (int)I_DIM;                                                 // dim0_stride lo32
  g1[6] = 0;
  g1[7] = 0;
  i32x4 gz4; gz4[0] = 0; gz4[1] = 0; gz4[2] = 0; gz4[3] = 0;
  i32x8 gz8;
  #pragma unroll
  for (int j = 0; j < 8; ++j) gz8[j] = 0;
  __builtin_amdgcn_tensor_load_to_lds(g0, g1, gz4, gz4, gz8, 0);
}

__global__ __launch_bounds__(256) void snn_lif_wmma_kernel(
    const float* __restrict__ x,      // [B, I, T]
    const float* __restrict__ W,      // [O, I]
    const float* __restrict__ bias,   // [O]
    float* __restrict__ out) {        // [B, O, T] spikes
  extern __shared__ char smem[];
  unsigned short* xs = (unsigned short*)smem;     // bf16 [t][k], padded rows
  unsigned*       xs32 = (unsigned*)smem;         // dword view for packed stores
  float* zb = (float*)(smem + Z_OFF);             // f32  [o][t], padded rows

  const int tid  = threadIdx.x;
  const int lane = tid & 31;
  const int wv   = tid >> 5;
  const int b    = blockIdx.y;
  const int o0   = blockIdx.x * OTILE;

  v8f acc[NTB];
  #pragma unroll
  for (int i = 0; i < NTB; ++i)
    #pragma unroll
    for (int r = 0; r < 8; ++r) acc[i][r] = 0.0f;

  const float* xg = x + ((size_t)b * I_DIM) * T_DIM;
  const float* wg = W + (size_t)o0 * I_DIM;
  const int arow = lane & 15;
  const int acol = (lane >> 4) << 3;     // A K-run start (halfwords)

  // ---- pipeline prologue: prefetch x chunk 0 into regs, TDM W chunk 0 ----
  f32x4 R[8];
  #pragma unroll
  for (int j = 0; j < 4; ++j) {
    int p  = j * 256 + tid;
    int ki = (p >> 6) << 1;
    int t  = (p & 63) << 2;
    R[2 * j + 0] = *(const f32x4*)(xg + (ki + 0) * T_DIM + t);
    R[2 * j + 1] = *(const f32x4*)(xg + (ki + 1) * T_DIM + t);
  }
  if (wv == 0) tdm_load_w(wg, (unsigned)(uintptr_t)(smem + WS_OFF));

  for (int kc = 0; kc < NKC; ++kc) {
    const float* wsb = (float*)(smem + WS_OFF + (kc & 1) * WS_BYTES);

    // ---- stage x chunk kc from regs -> xs bf16 [t][k], k-pairs packed ----
    #pragma unroll
    for (int j = 0; j < 4; ++j) {
      int p  = j * 256 + tid;
      int ki = (p >> 6) << 1;
      int t  = (p & 63) << 2;
      #pragma unroll
      for (int q = 0; q < 4; ++q)
        xs32[(t + q) * (XROW_H / 2) + (ki >> 1)] =
            pack_bf16x2(R[2 * j + 0][q], R[2 * j + 1][q]);
    }

    // ---- prefetch x chunk kc+1 into regs (latency hidden by WMMA burst) ----
    if (kc + 1 < NKC) {
      const float* xc = xg + (size_t)((kc + 1) * KC) * T_DIM;
      #pragma unroll
      for (int j = 0; j < 4; ++j) {
        int p  = j * 256 + tid;
        int ki = (p >> 6) << 1;
        int t  = (p & 63) << 2;
        R[2 * j + 0] = *(const f32x4*)(xc + (ki + 0) * T_DIM + t);
        R[2 * j + 1] = *(const f32x4*)(xc + (ki + 1) * T_DIM + t);
      }
    }

    if (wv == 0) __builtin_amdgcn_s_wait_tensorcnt(0);  // ws[kc&1] ready
    __syncthreads();                                    // xs + ws visible to all

    // ---- TDM-prefetch next W chunk into the other buffer ----
    if (kc + 1 < NKC && wv == 0)
      tdm_load_w(wg + (kc + 1) * KC,
                 (unsigned)(uintptr_t)(smem + WS_OFF + ((kc + 1) & 1) * WS_BYTES));

    // ---- B fragment: lane N = l%16, K = (l/16)*8 + 0..7, 16 + (l/16)*8 + 0..7 ----
    union { v16bf v; __bf16 h[16]; } bfr;
    {
      const float* wrow = wsb + (wv * 16 + (lane & 15)) * WROW_F + ((lane >> 4) << 3);
      #pragma unroll
      for (int j = 0; j < 16; ++j) bfr.h[j] = (__bf16)wrow[(j & 7) + ((j >> 3) << 4)];
    }

    // ---- preload ALL 16 A fragments (32 independent ds_load_b128) ----
    union AF { v16bf v; i32x4 q[2]; };
    AF af[NTB];
    #pragma unroll
    for (int tb = 0; tb < NTB; ++tb) {
      const unsigned short* ap = xs + (tb * 16 + arow) * XROW_H + acol;
      af[tb].q[0] = *(const i32x4*)(ap);        // K = (l/16)*8 + 0..7
      af[tb].q[1] = *(const i32x4*)(ap + 16);   // K = 16 + (l/16)*8 + 0..7
    }

    // Fence: keeps all A loads ahead of the WMMA burst (descending dscnt waits).
    __builtin_amdgcn_sched_barrier(0);

    // ---- 16 back-to-back WMMAs ----
    #pragma unroll
    for (int tb = 0; tb < NTB; ++tb)
      acc[tb] = __builtin_amdgcn_wmma_f32_16x16x32_bf16(
          false, af[tb].v, false, bfr.v, (short)0, acc[tb], false, false);

    __builtin_amdgcn_sched_barrier(0);
    __syncthreads();   // chunk consumed; next iteration may overwrite xs
  }

  // ---- spill z tiles to LDS [o][t]: contiguous 8-float runs per tile ----
  {
    float* zrow = zb + (wv * 16 + (lane & 15)) * ZROW_F + ((lane >> 4) << 3);
    #pragma unroll
    for (int tb = 0; tb < NTB; ++tb) {
      f32x4 lo = __builtin_shufflevector(acc[tb], acc[tb], 0, 1, 2, 3);
      f32x4 hi = __builtin_shufflevector(acc[tb], acc[tb], 4, 5, 6, 7);
      *(f32x4*)(zrow + tb * 16 + 0) = lo;
      *(f32x4*)(zrow + tb * 16 + 4) = hi;
    }
  }
  __syncthreads();

  // ---- leaky integrate-and-fire scan: one thread per output channel ----
  if (tid < OTILE) {
    const int o = o0 + tid;
    const float bb = bias[o];
    const float* zrow = zb + tid * ZROW_F;
    float* op = out + ((size_t)b * O_DIM + o) * T_DIM;
    float mem = 0.0f;
    for (int t = 0; t < T_DIM; t += 4) {
      f32x4 zv = *(const f32x4*)(zrow + t);       // ds_load_b128
      f32x4 s4;
      #pragma unroll
      for (int j = 0; j < 4; ++j) {
        float z = zv[j] + bb;
        mem = 0.9f * mem + z;                     // leaky integrate
        float spk = (mem > 1.0f) ? 1.0f : 0.0f;   // threshold
        mem = (1.0f - spk) * mem;                 // reset
        s4[j] = spk;
      }
      *(f32x4*)(op + t) = s4;                     // coalesced 16B store
    }
  }
}

extern "C" void kernel_launch(void* const* d_in, const int* in_sizes, int n_in,
                              void* d_out, int out_size, void* d_ws, size_t ws_size,
                              hipStream_t stream) {
  (void)in_sizes; (void)n_in; (void)out_size; (void)d_ws; (void)ws_size;
  const float* x = (const float*)d_in[0];
  const float* W = (const float*)d_in[1];
  const float* b = (const float*)d_in[2];
  float* out = (float*)d_out;
  dim3 grid(O_DIM / OTILE, B_DIM);   // 8 o-tiles x 128 batches = 1024 WGs
  snn_lif_wmma_kernel<<<grid, 256, LDS_TOTAL, stream>>>(x, W, b, out);
}
